// VQEmbedding_66116726554650
// MI455X (gfx1250) — matrix operand order repaired and checked by reading, not compiled, not run on hardware
//
#include <hip/hip_runtime.h>

// ---------------------------------------------------------------------------
// VQ codebook nearest-neighbor + gather for MI455X (gfx1250, wave32, WMMA)
//   argmin_k ( ||c_k||^2 - 2 * x . c_k )  then  out = codebook[idx] (twice)
// GEMM done with v_wmma_f32_16x16x32_bf16 (f32 accumulate), argmin fused.
// B-operand LDS loads are double-buffered so ds_load latency overlaps WMMA.
// ---------------------------------------------------------------------------

typedef __attribute__((ext_vector_type(16))) __bf16 v16bf;
typedef __attribute__((ext_vector_type(8)))  float  v8f;

#define N_CODE     8192
#define EMB        256
#define NTOK       32768          // 32 * 1024
#define TOK_PER_WG 256            // 8 waves * 2 subtiles * 16 rows
#define SUBT       2
#define KTILES     (N_CODE / 16)  // 512 code tiles
#define BSTRIDE    264            // bf16 elems per code row in LDS (256 + 8 pad)

union V16 {
    unsigned short us[16];
    uint4          q[2];
    v16bf          v;
};

__device__ __forceinline__ unsigned short f2bf(float f) {
    unsigned u = __builtin_bit_cast(unsigned, f);
    // round-to-nearest-even f32 -> bf16
    return (unsigned short)((u + 0x7FFFu + ((u >> 16) & 1u)) >> 16);
}

__device__ __forceinline__ void pack8(V16& p, int o, float4 a, float4 b) {
    p.us[o + 0] = f2bf(a.x); p.us[o + 1] = f2bf(a.y);
    p.us[o + 2] = f2bf(a.z); p.us[o + 3] = f2bf(a.w);
    p.us[o + 4] = f2bf(b.x); p.us[o + 5] = f2bf(b.y);
    p.us[o + 6] = f2bf(b.z); p.us[o + 7] = f2bf(b.w);
}

// ---- Pre-pass: exact f32 ||c_k||^2 for all 8192 codes (one wave per row) ---
__global__ __launch_bounds__(256) void vq_csq_kernel(
    const float* __restrict__ cb, float* __restrict__ csq) {
    int g    = blockIdx.x * 256 + threadIdx.x;
    int row  = g >> 5;
    int lane = g & 31;
    const float* rp = cb + (size_t)row * EMB + lane * 8;
    float4 a = *(const float4*)rp;
    float4 b = *(const float4*)(rp + 4);
    float s = a.x * a.x + a.y * a.y + a.z * a.z + a.w * a.w +
              b.x * b.x + b.y * b.y + b.z * b.z + b.w * b.w;
#pragma unroll
    for (int m = 16; m >= 1; m >>= 1) s += __shfl_xor(s, m, 32);
    if (lane == 0) csq[row] = s;
}

// ---- Main fused GEMM + argmin + gather kernel ------------------------------
__global__ __launch_bounds__(256) void vq_main_kernel(
    const float* __restrict__ x, const float* __restrict__ cb,
    const float* __restrict__ csq, float* __restrict__ out) {
    __shared__ __align__(16) unsigned short Bsh[16][BSTRIDE];
    __shared__ int widx[TOK_PER_WG];

    const int tid  = threadIdx.x;
    const int lane = tid & 31;
    const int wave = tid >> 5;
    const int j    = lane & 15;   // code column in tile / token row in subtile
    const int h    = lane >> 4;   // wave half (WMMA layout)
    const int wgBase = blockIdx.x * TOK_PER_WG;

    // ---- Load A (2 token subtiles) into registers in WMMA A-layout --------
    // lane (j,h) of subtile s holds row (s*16+j), K runs [16h,16h+16)
    v16bf a[SUBT][8];
#pragma unroll
    for (int s = 0; s < SUBT; ++s) {
        const int row = wgBase + wave * (SUBT * 16) + s * 16 + j;
        const float* rp = x + (size_t)row * EMB;
#pragma unroll
        for (int k = 0; k < 8; ++k) {
            const int d0 = 32 * k + 16 * h;
            float4 f0 = *(const float4*)(rp + d0);
            float4 f1 = *(const float4*)(rp + d0 + 4);
            float4 f2 = *(const float4*)(rp + d0 + 8);
            float4 f3 = *(const float4*)(rp + d0 + 12);
            V16 p;
            pack8(p, 0, f0, f1);
            pack8(p, 8, f2, f3);
            a[s][k] = p.v;
        }
    }

    float bestd[SUBT][8];
    int   besti[SUBT][8];
#pragma unroll
    for (int s = 0; s < SUBT; ++s)
#pragma unroll
        for (int r = 0; r < 8; ++r) { bestd[s][r] = 3.402823466e38f; besti[s][r] = 0; }

    // ---- B staging map: thread -> (code row, 16-col span), coalesced ------
    const int brow = tid >> 4;
    const int bc0  = (tid & 15) * 16;

    // prefetch tile 0 into registers
    float4 r0, r1, r2, r3;
    {
        const float* bp = cb + (size_t)brow * EMB + bc0;
        r0 = *(const float4*)(bp + 0);
        r1 = *(const float4*)(bp + 4);
        r2 = *(const float4*)(bp + 8);
        r3 = *(const float4*)(bp + 12);
    }

    for (int t = 0; t < KTILES; ++t) {
        __syncthreads();                 // LDS free (previous tile consumed)
        {
            V16 p;
            pack8(p, 0, r0, r1);
            pack8(p, 8, r2, r3);
            *(uint4*)&Bsh[brow][bc0]     = p.q[0];
            *(uint4*)&Bsh[brow][bc0 + 8] = p.q[1];
        }
        __syncthreads();                 // staging visible

        const float cs = csq[t * 16 + j];     // exact ||c||^2 for this lane's code
        if (t + 1 < KTILES) {                 // prefetch next tile (overlaps WMMA)
            const float* bp = cb + ((size_t)((t + 1) * 16 + brow)) * EMB + bc0;
            r0 = *(const float4*)(bp + 0);
            r1 = *(const float4*)(bp + 4);
            r2 = *(const float4*)(bp + 8);
            r3 = *(const float4*)(bp + 12);
        }

        // ---- 8 k-steps, B operand double-buffered from LDS ----------------
        v8f acc0 = {}, acc1 = {};
        V16 bbuf[2];
        {
            const uint4* q = (const uint4*)&Bsh[j][16 * h];
            bbuf[0].q[0] = q[0];
            bbuf[0].q[1] = q[1];
        }
#pragma unroll
        for (int k = 0; k < 8; ++k) {
            if (k + 1 < 8) {   // issue next ds_load pair before this k's WMMAs
                const uint4* q = (const uint4*)&Bsh[j][32 * (k + 1) + 16 * h];
                bbuf[(k + 1) & 1].q[0] = q[0];
                bbuf[(k + 1) & 1].q[1] = q[1];
            }
            acc0 = __builtin_amdgcn_wmma_f32_16x16x32_bf16(
                false, a[0][k], false, bbuf[k & 1].v, (short)0, acc0, false, false);
            acc1 = __builtin_amdgcn_wmma_f32_16x16x32_bf16(
                false, a[1][k], false, bbuf[k & 1].v, (short)0, acc1, false, false);
        }

        const int code = t * 16 + j;
#pragma unroll
        for (int r = 0; r < 8; ++r) {
            float d0 = cs - 2.0f * acc0[r];
            float d1 = cs - 2.0f * acc1[r];
            if (d0 < bestd[0][r]) { bestd[0][r] = d0; besti[0][r] = code; }
            if (d1 < bestd[1][r]) { bestd[1][r] = d1; besti[1][r] = code; }
        }
    }

    // ---- Cross-lane argmin (16 code columns per half-wave) ----------------
#pragma unroll
    for (int s = 0; s < SUBT; ++s)
#pragma unroll
        for (int r = 0; r < 8; ++r) {
            float bd = bestd[s][r];
            int   bi = besti[s][r];
#pragma unroll
            for (int m = 1; m < 16; m <<= 1) {
                float od = __shfl_xor(bd, m, 32);
                int   oi = __shfl_xor(bi, m, 32);
                if (od < bd || (od == bd && oi < bi)) { bd = od; bi = oi; }
            }
            if (j == 0)  // C/D layout: VGPR r on half h is token row r + 8h
                widx[wave * (SUBT * 16) + s * 16 + r + 8 * h] = bi;
        }
    __syncthreads();

    // ---- Gather winning code rows to both outputs (coalesced) -------------
    const size_t TOTAL = (size_t)NTOK * EMB;
    for (int tkn = 0; tkn < TOK_PER_WG; ++tkn) {
        const int idx = widx[tkn];
        const float v = cb[(size_t)idx * EMB + tid];
        const size_t o = ((size_t)(wgBase + tkn)) * EMB + tid;
        out[o]         = v;   // z_q_x
        out[o + TOTAL] = v;   // z_q_x_bar
    }
}

extern "C" void kernel_launch(void* const* d_in, const int* in_sizes, int n_in,
                              void* d_out, int out_size, void* d_ws, size_t ws_size,
                              hipStream_t stream) {
    const float* z_e_x    = (const float*)d_in[0];
    const float* codebook = (const float*)d_in[1];
    float*       out      = (float*)d_out;
    float*       csq      = (float*)d_ws;   // 8192 floats of scratch

    // Pass 1: exact ||c_k||^2
    vq_csq_kernel<<<(N_CODE * 32) / 256, 256, 0, stream>>>(codebook, csq);
    // Pass 2: fused bf16-WMMA distance GEMM + argmin + gather
    vq_main_kernel<<<NTOK / TOK_PER_WG, 256, 0, stream>>>(z_e_x, codebook, csq, out);
}
